// SelfAttention_4913442586818
// MI455X (gfx1250) — matrix operand order repaired and checked by reading, not compiled, not run on hardware
//
#include <hip/hip_runtime.h>
#include <hip/hip_bf16.h>

// Problem constants (from reference): B=16, E=42, D=256
#define NB   16
#define EE   42
#define DD   256
#define NKEY (EE * EE)          // 1764 tokens
#define QT   ((NKEY + 15) / 16) // 111 query tiles of 16
#define KBLK ((NKEY + 31) / 32) // 56 key blocks of 32
#define QPITCH 264              // bf16 pitch for Q tile in LDS (bank-friendly)
#define PPITCH 40               // bf16 pitch for P^T tile rows (16B aligned)

typedef __attribute__((ext_vector_type(16))) __bf16 v16bf;
typedef __attribute__((ext_vector_type(2)))  __bf16 v2bf;
typedef __attribute__((ext_vector_type(2)))  float  v2f;
typedef __attribute__((ext_vector_type(8)))  float  v8f;

union BF16x16 {
    unsigned short u[16];
    unsigned int   w[8];
    uint4          q[2];
    v16bf          v;
};

// Pack two f32 into one dword of bf16 -> single v_cvt_pk_bf16_f32.
__device__ __forceinline__ unsigned int pack2bf(float a, float b) {
    v2f f; f.x = a; f.y = b;
    union { v2bf v; unsigned int u; } c;
    c.v = __builtin_convertvector(f, v2bf);
    return c.u;
}

// Two waves (64 threads) cooperatively handle one 16-query tile of one batch,
// flash-attention style.  Wave w owns 16 of each 32-key block for QK^T and
// 128 of the 256 output features for P*V -> no redundant WMMA work, and the
// O^T accumulator per wave is only 64 VGPRs, leaving the scheduler headroom
// to pipeline the streamed global loads.
__global__ __launch_bounds__(64, 1)
void attn_flash_kernel(const float* __restrict__ x,
                       const float* __restrict__ mem,
                       const unsigned char* __restrict__ mask,
                       const float* __restrict__ dot_scale,
                       float* __restrict__ out)
{
    __shared__ __align__(16) unsigned short q_lds[16 * QPITCH]; // Q tile, bf16
    __shared__ __align__(16) unsigned short p_lds[16 * PPITCH]; // P^T tile [q][key]
    __shared__ __align__(16) float st_tmax[2 * 16];  // per-wave tile max  [w][q]
    __shared__ __align__(16) float st_psum[2 * 16];  // per-wave tile sum  [w][q]
    __shared__ __align__(16) float st_alpha[16];     // rescale factor per q
    __shared__ __align__(16) float st_sum[16];       // final row sums per q

    const int tid  = threadIdx.x;    // 0..63
    const int w    = tid >> 5;       // wave id 0/1
    const int lane = tid & 31;
    const int l16  = lane & 15;
    const int hi   = (lane >> 4) & 1;
    const int qbase = blockIdx.x * 16;
    const int b     = blockIdx.y;

    const float*         xb = x   + (size_t)b * NKEY * DD;
    const float*         mb = mem + (size_t)b * NKEY * DD;
    const unsigned char* mk = mask + (size_t)b * NKEY;

    // ---- Stage Q = x * dot_scale into LDS as bf16 (64 threads cooperate)
    {
        const int row = tid & 15;
        const int qd4 = tid >> 4;                 // quarter of D
        int qr = qbase + row; if (qr >= NKEY) qr = NKEY - 1;
        const float* xr = xb + (size_t)qr * DD + qd4 * 64;
        unsigned short* qd = &q_lds[row * QPITCH + qd4 * 64];
        #pragma unroll
        for (int i = 0; i < 16; ++i) {
            float4 xv = ((const float4*)xr)[i];
            float4 sv = ((const float4*)dot_scale)[qd4 * 16 + i];
            uint2 o;
            o.x = pack2bf(xv.x * sv.x, xv.y * sv.y);
            o.y = pack2bf(xv.z * sv.z, xv.w * sv.w);
            *(uint2*)(qd + i * 4) = o;
        }
    }
    __syncthreads();

    // ---- flash state (identical in both waves): lanes<16 own rows 0..7
    float row_max[8], row_sum[8];
    #pragma unroll
    for (int r = 0; r < 8; ++r) { row_max[r] = -3.0e38f; row_sum[r] = 0.0f; }
    v8f acc[8];   // O^T half: acc[oc][r] = O[d=128*w+16*oc+r+hi*8][q=l16]
    #pragma unroll
    for (int oc = 0; oc < 8; ++oc) acc[oc] = (v8f){0,0,0,0,0,0,0,0};

    #pragma unroll 1
    for (int kb = 0; kb < KBLK; ++kb) {
        const int k0   = kb * 32;
        const int keyw = k0 + 16 * w + l16;          // this wave's key column
        const int ckw  = keyw < NKEY ? keyw : NKEY - 1;
        const float* mrow = mb + (size_t)ckw * DD + hi * 16;

        if (kb + 1 < KBLK)
            __builtin_prefetch(mrow + 32 * DD, 0, 3); // global_prefetch_b8

        // ---- S_w = Q * M_w^T   (one 16-key tile, 8 K-chunks of 32 over D)
        v8f s = (v8f){0,0,0,0,0,0,0,0};
        #pragma unroll
        for (int kc = 0; kc < 8; ++kc) {
            BF16x16 a, bt;
            const unsigned short* qrow = &q_lds[l16 * QPITCH + kc * 32 + hi * 16];
            a.q[0] = *(const uint4*)(qrow);
            a.q[1] = *(const uint4*)(qrow + 8);
            #pragma unroll
            for (int i = 0; i < 4; ++i) {            // 16 contiguous floats
                float4 v0 = ((const float4*)(mrow + kc * 32))[i];
                bt.w[2*i+0] = pack2bf(v0.x, v0.y);
                bt.w[2*i+1] = pack2bf(v0.z, v0.w);
            }
            s = __builtin_amdgcn_wmma_f32_16x16x32_bf16(false, a.v, false, bt.v,
                                                        (short)0, s, false, false);
        }

        // ---- mask + per-wave tile max (input_dot cancels -> w_lin unused)
        const float madd = (keyw < NKEY) ? (mk[keyw] ? 0.0f : -1.0e30f) : -3.0e38f;
        float tmax[8];
        #pragma unroll
        for (int r = 0; r < 8; ++r) {
            float v0 = s[r] + madd;
            s[r] = v0;
            float t = v0;
            t = fmaxf(t, __shfl_xor(t, 1));
            t = fmaxf(t, __shfl_xor(t, 2));
            t = fmaxf(t, __shfl_xor(t, 4));
            t = fmaxf(t, __shfl_xor(t, 8));          // within 16-lane half
            tmax[r] = t;
        }
        if (lane == 0) {
            #pragma unroll
            for (int r = 0; r < 8; ++r) st_tmax[w * 16 + r] = tmax[r];
        }
        if (lane == 16) {
            #pragma unroll
            for (int r = 0; r < 8; ++r) st_tmax[w * 16 + 8 + r] = tmax[r];
        }
        __syncthreads();   // barrier #1: exchange tile maxima across waves

        // ---- online softmax with the cross-wave max
        float alpha[8], psum[8];
        #pragma unroll
        for (int r = 0; r < 8; ++r) {
            const int q = r + 8 * hi;
            float tall = fmaxf(st_tmax[q], st_tmax[16 + q]);
            float nm   = fmaxf(row_max[r], tall);
            alpha[r]   = __expf(row_max[r] - nm);
            row_max[r] = nm;
            float p = __expf(s[r] - nm);
            s[r] = p;
            float ps = p;
            ps += __shfl_xor(ps, 1);
            ps += __shfl_xor(ps, 2);
            ps += __shfl_xor(ps, 4);
            ps += __shfl_xor(ps, 8);
            psum[r] = ps;
        }
        if (lane == 0) {
            #pragma unroll
            for (int r = 0; r < 8; ++r) st_psum[w * 16 + r] = psum[r];
            if (w == 0) {
                #pragma unroll
                for (int r = 0; r < 8; ++r) st_alpha[r] = alpha[r];
            }
        }
        if (lane == 16) {
            #pragma unroll
            for (int r = 0; r < 8; ++r) st_psum[w * 16 + 8 + r] = psum[r];
            if (w == 0) {
                #pragma unroll
                for (int r = 0; r < 8; ++r) st_alpha[8 + r] = alpha[r];
            }
        }
        // ---- P -> LDS already transposed, [q][key] bf16 (pitch PPITCH).
        //      8 independent b16 stores per lane; each half-wave writes one
        //      contiguous 32B row segment -> conflict-free, no RAW chain.
        {
            const int pb = 16 * w + l16;             // key column within block
            #pragma unroll
            for (int i = 0; i < 4; ++i) {
                unsigned int pv = pack2bf(s[2*i], s[2*i+1]);
                p_lds[(2*i     + 8 * hi) * PPITCH + pb] = (unsigned short)pv;
                p_lds[(2*i + 1 + 8 * hi) * PPITCH + pb] = (unsigned short)(pv >> 16);
            }
        }
        __syncthreads();   // barrier #2: P + partial sums visible

        // ---- combine partial sums; rescale accumulator
        #pragma unroll
        for (int r = 0; r < 8; ++r) {
            const int q = r + 8 * hi;
            row_sum[r] = row_sum[r] * alpha[r] + st_psum[q] + st_psum[16 + q];
        }
        const float sc = st_alpha[l16];
        if (__ballot(sc != 1.0f) != 0ull) {          // skip when max unchanged
            #pragma unroll
            for (int oc = 0; oc < 8; ++oc)
                acc[oc] = acc[oc] * sc;              // v_pk_mul_f32
        }

        // ---- B = P^T (32 keys x 16 queries): two aligned b128 LDS loads
        BF16x16 bp;
        {
            const unsigned short* pr = &p_lds[l16 * PPITCH + hi * 16];
            bp.q[0] = *(const uint4*)(pr);           // K = hi*16 + 0..7
            bp.q[1] = *(const uint4*)(pr + 8);       // K = hi*16 + 8..15
        }

        // ---- O^T += M^T * P^T : 8 WMMAs on this wave's 128-feature half
        int koff[16];
        #pragma unroll
        for (int j = 0; j < 16; ++j) {
            int key = k0 + j + hi * 16; // contiguous K idx (matches B)
            if (key >= NKEY) key = NKEY - 1;                   // p==0 there anyway
            koff[j] = key * DD;
        }
        #pragma unroll
        for (int oc = 0; oc < 8; ++oc) {
            BF16x16 am;
            const int dcol = 128 * w + oc * 16 + l16;
            #pragma unroll
            for (int j = 0; j < 8; ++j)
                am.w[j] = pack2bf(mb[koff[2*j]   + dcol],   // coalesced across lanes
                                  mb[koff[2*j+1] + dcol]);
            acc[oc] = __builtin_amdgcn_wmma_f32_16x16x32_bf16(false, am.v, false, bp.v,
                                                              (short)0, acc[oc], false, false);
        }
    }

    // ---- finalize: divide by row sums, store this wave's O half
    if (w == 0 && lane == 0) {
        #pragma unroll
        for (int r = 0; r < 8; ++r) st_sum[r] = row_sum[r];
    }
    if (w == 0 && lane == 16) {
        #pragma unroll
        for (int r = 0; r < 8; ++r) st_sum[8 + r] = row_sum[r];
    }
    __syncthreads();
    const float inv = 1.0f / st_sum[l16];
    const int qg = qbase + l16;
    if (qg < NKEY) {
        float* orow = out + ((size_t)b * NKEY + qg) * (2 * DD) + DD + 128 * w;
        #pragma unroll
        for (int oc = 0; oc < 8; ++oc) {
            v8f sc8 = acc[oc] * inv;                 // v_pk_mul_f32
            float4 lo, hi4;
            lo.x  = sc8[0]; lo.y  = sc8[1]; lo.z  = sc8[2]; lo.w  = sc8[3];
            hi4.x = sc8[4]; hi4.y = sc8[5]; hi4.z = sc8[6]; hi4.w = sc8[7];
            *(float4*)(orow + oc * 16 + hi * 8)     = lo;   // d contiguous
            *(float4*)(orow + oc * 16 + hi * 8 + 4) = hi4;
        }
    }
}

// Copy x into the first half of every output row (concat([x, O], -1)).
__global__ void copy_x_kernel(const float* __restrict__ x, float* __restrict__ out)
{
    size_t t = (size_t)blockIdx.x * blockDim.x + threadIdx.x;
    const size_t total = (size_t)NB * NKEY * (DD / 4);
    if (t >= total) return;
    size_t row = t / (DD / 4);
    int    j   = (int)(t % (DD / 4));
    float4 v = ((const float4*)x)[row * (DD / 4) + j];
    ((float4*)(out + row * (size_t)(2 * DD)))[j] = v;
}

extern "C" void kernel_launch(void* const* d_in, const int* in_sizes, int n_in,
                              void* d_out, int out_size, void* d_ws, size_t ws_size,
                              hipStream_t stream)
{
    (void)in_sizes; (void)n_in; (void)out_size; (void)d_ws; (void)ws_size;
    const float*         x         = (const float*)d_in[0];
    const float*         mem       = (const float*)d_in[1];
    const unsigned char* mask      = (const unsigned char*)d_in[2]; // jax bool = 1 byte
    // d_in[3] = w_lin: provably unused (per-row constant cancels in softmax)
    const float*         dot_scale = (const float*)d_in[4];
    float*               out       = (float*)d_out;

    dim3 grid(QT, NB);
    hipLaunchKernelGGL(attn_flash_kernel, grid, dim3(64), 0, stream,
                       x, mem, mask, dot_scale, out);

    const size_t total = (size_t)NB * NKEY * (DD / 4);
    const int thr = 256;
    const int blk = (int)((total + thr - 1) / thr);
    hipLaunchKernelGGL(copy_x_kernel, dim3(blk), dim3(thr), 0, stream, x, out);
}
